// LSNN_27273042329871
// MI455X (gfx1250) — matrix operand order repaired and checked by reading, not compiled, not run on hardware
//
#include <hip/hip_runtime.h>
#include <hip/hip_bf16.h>
#include <stdint.h>

#define B_ 64
#define T_ 1024
#define I_ 128
#define H_ 512
#define O_ 10
#define NWG 16
#define WGSIZE 256
#define ALPHA_ 0.9512294245007140f  // exp(-1/20)

typedef __attribute__((ext_vector_type(16))) __bf16 v16bf;
typedef __attribute__((ext_vector_type(8)))  float  v8f;
typedef __attribute__((ext_vector_type(4)))  unsigned int u32x4;
typedef __attribute__((ext_vector_type(8)))  int     i32x8;
typedef __attribute__((ext_vector_type(4)))  int     i32x4;

union V16 {
  v16bf v;
  uint4 q[2];
  unsigned int u[8];
  unsigned short us[16];
};

// Native fp32 -> bf16 (RNE) via hardware cvt, as raw bits.
__device__ __forceinline__ unsigned short bfb(float f) {
  union { __bf16 h; unsigned short u; } cv;
  cv.h = (__bf16)f;
  return cv.u;
}
__device__ __forceinline__ unsigned pk_bf16(float a, float b) {
  union { __bf16 h[2]; unsigned u; } cv;
  cv.h[0] = (__bf16)a;
  cv.h[1] = (__bf16)b;
  return cv.u;
}

// Prep region sizes
#define R_WR   (H_ * H_)                    // 262144  pack Wr
#define R_WI   (I_ * H_)                    // 65536   pack Wi^T
#define R_SPK  (B_ * H_)                    // 32768   zero spikes(t=-1)
#define R_BAR  1                            //         reset barrier
#define R_X    (B_ * T_ * I_ / 4)           // 2097152 convert x (4 elems/thread)
#define PREP_TOTAL (R_WR + R_WI + R_SPK + R_BAR + R_X)

// ---------------------------------------------------------------------------
// Prep: pack Wr and Wi^T into the CDNA5 WMMA B-operand lane layout (bf16),
// convert x to bf16 (row-contiguous, directly loadable as A-operand),
// zero the t=0 spike buffer, reset the global barrier counter.
// B-layout (16x16x32 bf16): tile (kc,nt) of 32(K)x16(N); lane L (half=L/16,
// n=L%16): element s packs K = kc*32 + 8*half + (s&7) + 16*(s>>3), column N = nt*16+n.
// Tile stored as 32 lanes x 16 bf16 contiguous (lane offset L*16).
// ---------------------------------------------------------------------------
__global__ void snn_prep(const float* __restrict__ Wr, const float* __restrict__ Wi,
                         const float* __restrict__ x,
                         unsigned short* __restrict__ WrP, unsigned short* __restrict__ WiP,
                         unsigned short* __restrict__ xbf,
                         unsigned short* __restrict__ spk0, int* __restrict__ bar) {
  int tid = blockIdx.x * blockDim.x + threadIdx.x;
  if (tid < R_WR) {
    int tile = tid >> 9, within = tid & 511;
    int L = within >> 4, idx = within & 15;
    int j = idx >> 1, p = idx & 1;
    int hlf = L >> 4, n = L & 15;
    int kc = tile & 15, nt = tile >> 4;            // tileIdx = nt*16 + kc
    int K = kc * 32 + 8 * hlf + (idx & 7) + 16 * (idx >> 3);
    int N = nt * 16 + n;
    WrP[tid] = bfb(Wr[K * H_ + N]);                // B[k][n] = Wr[k,n]
  } else if (tid < R_WR + R_WI) {
    int t2 = tid - R_WR;
    int tile = t2 >> 9, within = t2 & 511;
    int L = within >> 4, idx = within & 15;
    int j = idx >> 1, p = idx & 1;
    int hlf = L >> 4, n = L & 15;
    int kc = tile & 3, nt = tile >> 2;             // tileIdx = nt*4 + kc
    int K = kc * 32 + 8 * hlf + (idx & 7) + 16 * (idx >> 3);
    int N = nt * 16 + n;
    WiP[t2] = bfb(Wi[N * I_ + K]);                 // B[k=i][n=h] = Wi[h,i]
  } else if (tid < R_WR + R_WI + R_SPK) {
    spk0[tid - (R_WR + R_WI)] = 0;                 // spikes(t=-1) = 0
  } else if (tid == R_WR + R_WI + R_SPK) {
    *bar = 0;
  } else if (tid < PREP_TOTAL) {
    // x fp32 -> bf16, 4 elements per thread, vectorized load/store
    size_t t4 = (size_t)(tid - (R_WR + R_WI + R_SPK + R_BAR));
    float4 f = *(const float4*)(x + t4 * 4);
    uint2 o;
    o.x = pk_bf16(f.x, f.y);
    o.y = pk_bf16(f.z, f.w);
    *(uint2*)(xbf + t4 * 4) = o;
  }
}

// ---------------------------------------------------------------------------
// Tensor Data Mover: 1-D slab copy global -> LDS (data_size = 4B dwords).
// ---------------------------------------------------------------------------
#if __has_builtin(__builtin_amdgcn_tensor_load_to_lds)
#define USE_TDM 1
__device__ __forceinline__ void tdm_load_lin(unsigned lds_off, const void* gp, unsigned ndw) {
  unsigned long long ga = (unsigned long long)(uintptr_t)gp;
  u32x4 g0;
  g0[0] = 1u;                                          // count=1, user mode
  g0[1] = lds_off;                                     // lds_addr (bytes)
  g0[2] = (unsigned)ga;                                // global_addr[31:0]
  g0[3] = (unsigned)((ga >> 32) & 0x01FFFFFFull) | (2u << 30);  // addr[56:32] | type=2
  i32x8 g1;
  g1[0] = (2 << 16);                                   // data_size = 4B
  g1[1] = (int)(ndw << 16);                            // tensor_dim0 lo16 (bits 79:48)
  g1[2] = (1 << 16);                                   // tensor_dim0 hi=0, tensor_dim1=1
  g1[3] = (int)(ndw << 16);                            // tile_dim0 (bits 127:112)
  g1[4] = 1;                                           // tile_dim1 = 1, tile_dim2 = 0
  g1[5] = (int)ndw;                                    // tensor_dim0_stride lo32
  g1[6] = 0;
  g1[7] = 0;
  i32x4 z4 = {0, 0, 0, 0};
#if __clang_major__ >= 23
  i32x8 z8 = {0, 0, 0, 0, 0, 0, 0, 0};
  __builtin_amdgcn_tensor_load_to_lds(g0, g1, z4, z4, z8, 0);
#else
  __builtin_amdgcn_tensor_load_to_lds(g0, g1, z4, z4, 0);
#endif
}
#endif

// ---------------------------------------------------------------------------
// Persistent LIF scan. 16 WGs x 256 threads. WG g owns H columns
// [32g, 32g+32); wave w owns rows [16*(w>>1), +16) and 16 columns
// (ntl = w&1). Membrane h and spike-rate sum stay in VGPRs for all 1024
// steps. Per step: 4 WMMA (input proj, K=128) + 16 WMMA (recurrent, K=512).
// All A/B operands are pure b128 loads (bf16 pre-converted/packed by prep).
// ---------------------------------------------------------------------------
__global__ void __launch_bounds__(WGSIZE)
snn_scan(const unsigned short* __restrict__ xbf, const float* __restrict__ bi,
         const unsigned short* __restrict__ WrP, const unsigned short* __restrict__ WiP,
         unsigned short* __restrict__ spkbuf, float* __restrict__ meanb,
         int* __restrict__ bar, float* __restrict__ hid) {
  __shared__ unsigned short lds_wr[16 * 1024];  // 32 KB: 2 nt x 16 kc x 512
  __shared__ unsigned short lds_wi[4 * 1024];   //  8 KB: 2 nt x  4 kc x 512

  const int g   = blockIdx.x;
  const int tid = threadIdx.x;
  const int w   = tid >> 5;
  const int L   = tid & 31;
  const int hlf = L >> 4;
  const int ln  = L & 15;
  const int M0  = (w >> 1) << 4;          // batch-row tile base
  const int ntl = w & 1;                  // local N tile
  const int col = g * 32 + ntl * 16 + ln; // hidden column this lane owns

  // Stage this WG's packed Wr/Wi slabs into LDS (TDM if available).
#ifdef USE_TDM
  if (w == 0) {
    tdm_load_lin((unsigned)(uintptr_t)&lds_wr[0], WrP + g * 16384, 8192);
    tdm_load_lin((unsigned)(uintptr_t)&lds_wi[0], WiP + g * 4096, 2048);
    __builtin_amdgcn_s_wait_tensorcnt(0);
  }
  __syncthreads();
#else
  for (int i = tid; i < 16384; i += WGSIZE) lds_wr[i] = WrP[g * 16384 + i];
  for (int i = tid; i < 4096;  i += WGSIZE) lds_wi[i] = WiP[g * 4096 + i];
  __syncthreads();
#endif

  const float bias = bi[col];
  v8f h    = {0.f, 0.f, 0.f, 0.f, 0.f, 0.f, 0.f, 0.f};
  v8f ssum = {0.f, 0.f, 0.f, 0.f, 0.f, 0.f, 0.f, 0.f};

  for (int t = 0; t < T_; ++t) {
    const unsigned short* spk  = spkbuf + (t & 1) * (B_ * H_);
    unsigned short* spknx      = spkbuf + ((t + 1) & 1) * (B_ * H_);
    v8f acc = {0.f, 0.f, 0.f, 0.f, 0.f, 0.f, 0.f, 0.f};

    // ---- input projection: acc += xbf[:,t,:] @ WiP, K = 128 ----
    const unsigned short* xrow = xbf + ((size_t)(M0 + ln) * T_ + t) * I_;
    __builtin_prefetch(xrow + I_, 0, 1);  // next timestep -> global_prefetch_b8
#pragma unroll
    for (int kc = 0; kc < 4; ++kc) {
      const int k0 = kc * 32 + 8 * hlf;
      V16 a;
      a.q[0] = *(const uint4*)(xrow + k0);        // K = k0..k0+7
      a.q[1] = *(const uint4*)(xrow + k0 + 16);   // K = k0+16..k0+23
      const v16bf* bmat = (const v16bf*)&lds_wi[((ntl * 4 + kc) * 512) + L * 16];
      acc = __builtin_amdgcn_wmma_f32_16x16x32_bf16(false, a.v, false, *bmat,
                                                    (short)0, acc, false, false);
    }

    // ---- recurrent: acc += spikes_prev @ WrP, K = 512 ----
    const unsigned short* srow = spk + (M0 + ln) * H_;
#pragma unroll
    for (int kc = 0; kc < 16; ++kc) {
      const int k0 = kc * 32 + 8 * hlf;
      V16 a;
      a.q[0] = *(const uint4*)(srow + k0);        // K = k0..k0+7
      a.q[1] = *(const uint4*)(srow + k0 + 16);   // K = k0+16..k0+23
      const v16bf* bmat = (const v16bf*)&lds_wr[((ntl * 16 + kc) * 512) + L * 16];
      acc = __builtin_amdgcn_wmma_f32_16x16x32_bf16(false, a.v, false, *bmat,
                                                    (short)0, acc, false, false);
    }

    // ---- LIF: leak + integrate, threshold, reset; emit spikes ----
#pragma unroll
    for (int r = 0; r < 8; ++r) {
      const int brow = M0 + r + 8 * hlf;         // C-layout: VGPR r -> M = r + 8*half
      float hv = ALPHA_ * h[r] + acc[r] + bias;
      float s  = (hv >= 1.0f) ? 1.0f : 0.0f;
      h[r]     = (s != 0.0f) ? 0.0f : hv;        // V_RESET = 0
      ssum[r] += s;
      hid[(size_t)brow * (T_ * H_) + (size_t)t * H_ + col] = s;   // fp32 spikes out
      spknx[brow * H_ + col] = (s != 0.0f) ? (unsigned short)0x3F80u
                                           : (unsigned short)0;   // bf16 for next GEMM
    }

    // ---- device-wide step barrier (monotonic counter) ----
    __syncthreads();
    if (tid == 0) {
      __hip_atomic_fetch_add(bar, 1, __ATOMIC_RELEASE, __HIP_MEMORY_SCOPE_AGENT);
      const int target = NWG * (t + 1);
      while (__hip_atomic_load(bar, __ATOMIC_ACQUIRE, __HIP_MEMORY_SCOPE_AGENT) < target)
        __builtin_amdgcn_s_sleep(2);
    }
    __syncthreads();
    __threadfence();
  }

  // rate readout input: mean over T
#pragma unroll
  for (int r = 0; r < 8; ++r) {
    const int brow = M0 + r + 8 * hlf;
    meanb[brow * H_ + col] = ssum[r] * (1.0f / (float)T_);
  }
}

// ---------------------------------------------------------------------------
// Readout: out[b,o] = mean[b,:] . Wo[o,:] + bo[o]   (640 dot products)
// ---------------------------------------------------------------------------
__global__ void snn_readout(const float* __restrict__ meanb, const float* __restrict__ Wo,
                            const float* __restrict__ bo, float* __restrict__ out) {
  int id = blockIdx.x * blockDim.x + threadIdx.x;
  if (id >= B_ * O_) return;
  int b = id / O_, o = id % O_;
  const float* m  = meanb + b * H_;
  const float* wo = Wo + o * H_;
  float s = bo[o];
  for (int k = 0; k < H_; ++k) s += m[k] * wo[k];
  out[id] = s;
}

extern "C" void kernel_launch(void* const* d_in, const int* in_sizes, int n_in,
                              void* d_out, int out_size, void* d_ws, size_t ws_size,
                              hipStream_t stream) {
  (void)in_sizes; (void)n_in; (void)out_size; (void)ws_size;
  const float* x  = (const float*)d_in[0];
  const float* Wi = (const float*)d_in[1];
  const float* bi = (const float*)d_in[2];
  const float* Wr = (const float*)d_in[3];
  const float* Wo = (const float*)d_in[4];
  const float* bo = (const float*)d_in[5];
  float* out = (float*)d_out;

  char* ws = (char*)d_ws;
  unsigned short* WrP    = (unsigned short*)(ws + 0);        // 512 KB packed Wr bf16
  unsigned short* WiP    = (unsigned short*)(ws + 524288);   // 128 KB packed Wi^T bf16
  unsigned short* spkbuf = (unsigned short*)(ws + 655360);   // 2 x 64 KB spike ping-pong
  float*          meanb  = (float*)(ws + 786432);            // 128 KB mean spikes
  int*            bar    = (int*)(ws + 917504);              // barrier counter
  unsigned short* xbf    = (unsigned short*)(ws + 1048576);  // 16.7 MB x in bf16

  const int prep_blocks = (PREP_TOTAL + 255) / 256;

  snn_prep<<<prep_blocks, 256, 0, stream>>>(Wr, Wi, x, WrP, WiP, xbf, spkbuf, bar);
  snn_scan<<<NWG, WGSIZE, 0, stream>>>(xbf, bi, WrP, WiP, spkbuf, meanb, bar,
                                       out + B_ * O_);
  snn_readout<<<(B_ * O_ + 255) / 256, 256, 0, stream>>>(meanb, Wo, bo, out);
}
